// nmODEAutoencoder_80358838108399
// MI455X (gfx1250) — matrix operand order repaired and checked by reading, not compiled, not run on hardware
//
#include <hip/hip_runtime.h>
#include <hip/hip_bf16.h>

typedef __attribute__((ext_vector_type(16))) _Float16 v16h;
typedef __attribute__((ext_vector_type(8)))  _Float16 v8h;
typedef __attribute__((ext_vector_type(4)))  _Float16 v4h;
typedef __attribute__((ext_vector_type(8)))  float    v8f;

#define BATCH   4096
#define IN_DIM  3072
#define HID     1024
#define NSTEPS  40

// Block tile 128x128, K-tile 32, 256 threads = 8 waves (2 in M x 4 in N).
// Each wave: 64x32 output = 4x2 tiles of 16x16 (8 x v8f accumulators).
#define BM 128
#define BN 128
#define BK 32
#define LDSS 40   // padded f16 row stride (32 data + 8 pad) -> 80B rows, bank-spread

__device__ __forceinline__ float fdyn(float y, float g) {
    float s = __sinf(y + g);
    return s * s - y;
}

// C[M,N] = A[M,K] @ W[N,K]^T   (both row-major, K contiguous: "TN" GEMM)
// MODE 0: epilogue = +bias, 40-step RK4 nmODE integration, store hidden
// MODE 1: epilogue = +bias, sigmoid, store output
template<int MODE>
__global__ __launch_bounds__(256)
void fused_gemm_kernel(const float* __restrict__ A,
                       const float* __restrict__ W,
                       const float* __restrict__ bias,
                       float* __restrict__ out,
                       int M, int N, int K)
{
    // double-buffered LDS tiles (2 x (10240 + 10240) B = 40 KB)
    __shared__ __align__(16) _Float16 lA[2][BM * LDSS];
    __shared__ __align__(16) _Float16 lB[2][BN * LDSS];

    const int tid  = threadIdx.x;
    const int lane = tid & 31;
    const int wave = tid >> 5;
    const int wm   = (wave >> 2) * 64;   // wave row offset inside block tile
    const int wn   = (wave & 3) * 32;    // wave col offset inside block tile
    const int rsel = lane & 15;
    const int hsel = lane >> 4;

    const int blockRow = blockIdx.y * BM;
    const int blockCol = blockIdx.x * BN;

    // staging assignment: 8 threads x float4 = 32 floats per row,
    // 32 rows per sweep, 4 sweeps to cover 128 rows.
    const int lr = tid >> 3;          // 0..31
    const int lc = (tid & 7) * 4;     // 0,4,...,28

    const float* pa = A + (size_t)(blockRow + lr) * K + lc;
    const float* pw = W + (size_t)(blockCol + lr) * K + lc;
    const size_t rstep = (size_t)32 * K;

    v8f acc[4][2];
    const v8f vzero = {0.f, 0.f, 0.f, 0.f, 0.f, 0.f, 0.f, 0.f};
    #pragma unroll
    for (int i = 0; i < 4; ++i)
        #pragma unroll
        for (int j = 0; j < 2; ++j)
            acc[i][j] = vzero;

    const int ntiles = K / BK;

    // ---- prologue: stage tile 0 into buffer 0 ----
    {
        float4 aR[4], wR[4];
        #pragma unroll
        for (int i = 0; i < 4; ++i) {   // issue all 8 loads, no intervening use
            aR[i] = *reinterpret_cast<const float4*>(pa + i * rstep);
            wR[i] = *reinterpret_cast<const float4*>(pw + i * rstep);
        }
        #pragma unroll
        for (int i = 0; i < 4; ++i) {
            v4h ah = { (_Float16)aR[i].x, (_Float16)aR[i].y,
                       (_Float16)aR[i].z, (_Float16)aR[i].w };
            *reinterpret_cast<v4h*>(&lA[0][(lr + 32 * i) * LDSS + lc]) = ah;
            v4h wh = { (_Float16)wR[i].x, (_Float16)wR[i].y,
                       (_Float16)wR[i].z, (_Float16)wR[i].w };
            *reinterpret_cast<v4h*>(&lB[0][(lr + 32 * i) * LDSS + lc]) = wh;
        }
    }
    __syncthreads();

    for (int kt = 0; kt < ntiles; ++kt) {
        const int cur = kt & 1;
        const int nxt = cur ^ 1;
        const bool has_next = (kt + 1) < ntiles;

        // ---- issue next tile's global loads (wait lands after the WMMAs) ----
        float4 aR[4], wR[4];
        if (has_next) {
            const size_t ko = (size_t)(kt + 1) * BK;
            #pragma unroll
            for (int i = 0; i < 4; ++i) {
                aR[i] = *reinterpret_cast<const float4*>(pa + i * rstep + ko);
                wR[i] = *reinterpret_cast<const float4*>(pw + i * rstep + ko);
            }
        }

        // ---- build WMMA fragments from LDS (current buffer) ----
        // A (16x32 f16): lanes 0-15 hold M=lane, K {0..7, 16..23};
        //                lanes 16-31 hold M=lane-16, K {8..15, 24..31}.
        const _Float16* curA = lA[cur];
        const _Float16* curB = lB[cur];
        v16h af[4];
        #pragma unroll
        for (int ti = 0; ti < 4; ++ti) {
            const _Float16* p = &curA[(wm + ti * 16 + rsel) * LDSS + hsel * 8];
            v8h lo = *reinterpret_cast<const v8h*>(p);        // K 0..7  / 8..15
            v8h hi = *reinterpret_cast<const v8h*>(p + 16);   // K 16..23 / 24..31
            af[ti] = __builtin_shufflevector(lo, hi,
                0,1,2,3,4,5,6,7,8,9,10,11,12,13,14,15);
        }
        // B (32x16 f16): lane holds column N=rsel; lanes 0-15 K {0..7, 16..23},
        //                lanes 16-31 K {8..15, 24..31}.
        v16h bf[2];
        #pragma unroll
        for (int tj = 0; tj < 2; ++tj) {
            const _Float16* p = &curB[(wn + tj * 16 + rsel) * LDSS + hsel * 8];
            v8h lo = *reinterpret_cast<const v8h*>(p);        // K 0..7  / 8..15
            v8h hi = *reinterpret_cast<const v8h*>(p + 16);   // K 16..23 / 24..31
            bf[tj] = __builtin_shufflevector(lo, hi,
                0,1,2,3,4,5,6,7,8,9,10,11,12,13,14,15);
        }

        // ---- 8 WMMAs per wave per K-tile ----
        #pragma unroll
        for (int ti = 0; ti < 4; ++ti)
            #pragma unroll
            for (int tj = 0; tj < 2; ++tj)
                acc[ti][tj] = __builtin_amdgcn_wmma_f32_16x16x32_f16(
                    false, af[ti], false, bf[tj],
                    (short)0, acc[ti][tj], false, false);

        // ---- convert + store next tile into the other buffer ----
        if (has_next) {
            _Float16* nA = lA[nxt];
            _Float16* nB = lB[nxt];
            #pragma unroll
            for (int i = 0; i < 4; ++i) {
                v4h ah = { (_Float16)aR[i].x, (_Float16)aR[i].y,
                           (_Float16)aR[i].z, (_Float16)aR[i].w };
                *reinterpret_cast<v4h*>(&nA[(lr + 32 * i) * LDSS + lc]) = ah;
                v4h wh = { (_Float16)wR[i].x, (_Float16)wR[i].y,
                           (_Float16)wR[i].z, (_Float16)wR[i].w };
                *reinterpret_cast<v4h*>(&nB[(lr + 32 * i) * LDSS + lc]) = wh;
            }
        }
        __syncthreads();
    }

    // ---- epilogue ----
    // C/D layout: lanes 0-15: VGPR j -> M=j, N=lane; lanes 16-31: M=8+j, N=lane-16.
    const float dt = 1.0f / NSTEPS;
    #pragma unroll
    for (int ti = 0; ti < 4; ++ti) {
        #pragma unroll
        for (int tj = 0; tj < 2; ++tj) {
            const int col = blockCol + wn + tj * 16 + rsel;
            const float bv = bias[col];
            if (MODE == 0) {
                // fused nmODE RK4, 8-wide ILP over the accumulator rows
                float g[8], y[8];
                #pragma unroll
                for (int j = 0; j < 8; ++j) { g[j] = acc[ti][tj][j] + bv; y[j] = 0.0f; }
                for (int s = 0; s < NSTEPS; ++s) {
                    #pragma unroll
                    for (int j = 0; j < 8; ++j) {
                        float k1 = fdyn(y[j], g[j]);
                        float k2 = fdyn(fmaf(0.5f * dt, k1, y[j]), g[j]);
                        float k3 = fdyn(fmaf(0.5f * dt, k2, y[j]), g[j]);
                        float k4 = fdyn(fmaf(dt, k3, y[j]), g[j]);
                        y[j] = fmaf(dt * (1.0f / 6.0f),
                                    k1 + 2.0f * (k2 + k3) + k4, y[j]);
                    }
                }
                #pragma unroll
                for (int j = 0; j < 8; ++j) {
                    const int row = blockRow + wm + ti * 16 + j + hsel * 8;
                    out[(size_t)row * N + col] = y[j];
                }
            } else {
                #pragma unroll
                for (int j = 0; j < 8; ++j) {
                    const int row = blockRow + wm + ti * 16 + j + hsel * 8;
                    const float z = acc[ti][tj][j] + bv;
                    out[(size_t)row * N + col] = 1.0f / (1.0f + __expf(-z));
                }
            }
        }
    }
}

extern "C" void kernel_launch(void* const* d_in, const int* in_sizes, int n_in,
                              void* d_out, int out_size, void* d_ws, size_t ws_size,
                              hipStream_t stream)
{
    (void)in_sizes; (void)n_in; (void)out_size; (void)d_ws; (void)ws_size;

    const float* x  = (const float*)d_in[0];   // [4096, 3072]
    const float* We = (const float*)d_in[1];   // [1024, 3072]
    const float* be = (const float*)d_in[2];   // [1024]
    const float* Wd = (const float*)d_in[3];   // [3072, 1024]
    const float* bd = (const float*)d_in[4];   // [3072]

    float* out = (float*)d_out;                          // [4096, 3072]
    float* hid = out + (size_t)BATCH * IN_DIM;           // [4096, 1024]

    dim3 blk(256);

    // Encoder GEMM + fused nmODE integration -> hidden
    dim3 g1(HID / BN, BATCH / BM);      // (8, 32)
    fused_gemm_kernel<0><<<g1, blk, 0, stream>>>(x, We, be, hid,
                                                 BATCH, HID, IN_DIM);

    // Decoder GEMM + fused sigmoid -> output
    dim3 g2(IN_DIM / BN, BATCH / BM);   // (24, 32)
    fused_gemm_kernel<1><<<g2, blk, 0, stream>>>(hid, Wd, bd, out,
                                                 BATCH, IN_DIM, HID);
}
